// Attention_8933531976242
// MI455X (gfx1250) — compile-verified, matrix operand order checked by reading, not run on hardware
//
#include <hip/hip_runtime.h>
#include <hip/hip_bf16.h>
#include <math.h>

typedef __attribute__((ext_vector_type(16))) __bf16 v16bf;
typedef __attribute__((ext_vector_type(8)))  __bf16 v8bf;
typedef __attribute__((ext_vector_type(4)))  __bf16 v4bf;
typedef __attribute__((ext_vector_type(8)))  float  v8f;
typedef __attribute__((ext_vector_type(4)))  float  v4f;

#define T_DIM 2048
#define B_DIM 2
#define E_DIM 1024
#define H_DIM 16
#define DH_DIM 64
#define NEGF (-3.4028234663852886e38f)
#define QSCALE 0.125f   // DH^-0.5

// ---------------------------------------------------------------------------
// Fragment helpers (gfx1250 wave32 WMMA layouts, CDNA5 ISA §7.12.2)
// A-matrix 16x32 bf16: lane m = lane&15; half = lane>>4 selects K offset
//   VGPR v (v<4): K = 2v,2v+1 (+8 if half) ; (v>=4): K = 16+2(v-4) (+8 if half)
// C/D 16x16 f32: col n = lane&15; row m = 8*(lane>>4) + vgpr_index
// B operand = column-major mirror of A => loading row-major [NxK] with the
// A loader computes X * Y^T, which is how every GEMM below is phrased.
// ---------------------------------------------------------------------------
__device__ __forceinline__ v16bf load_frag(const __bf16* p, int ld) {
  const int lane = threadIdx.x & 31;
  const int m = lane & 15;
  const int half = lane >> 4;
  const __bf16* row = p + m * ld;
  v16bf a;
#pragma unroll
  for (int v = 0; v < 8; ++v) {
    int kb = (v < 4) ? (2 * v + 8 * half) : (16 + 2 * (v - 4) + 8 * half);
    a[2 * v]     = row[kb];
    a[2 * v + 1] = row[kb + 1];
  }
  return a;
}

__device__ __forceinline__ v8f wmma_bf16(v16bf a, v16bf b, v8f c) {
  return __builtin_amdgcn_wmma_f32_16x16x32_bf16(false, a, false, b,
                                                 (short)0, c, false, false);
}

__device__ __forceinline__ v4bf cvt4(v4f x) {
  v4bf h;
#pragma unroll
  for (int e = 0; e < 4; ++e) h[e] = (__bf16)x[e];
  return h;
}

// ---------------------------------------------------------------------------
// Kernel 1: fused QKV projection.  X[4096,1024] (fp32, row r = t*B+b)
//   out = X * W^T + bias ; q scaled by QSCALE.  Writes bf16:
//   q,k -> [b][h][t][d] ; v -> transposed [b][h][d][t] (PV B-operand layout)
// Block tile 64x128, 8 waves of 32x32.  Software-pipelined: chunk k+1's
// global loads issue before chunk k's WMMAs, hiding HBM/L2 latency.
// ---------------------------------------------------------------------------
__global__ __launch_bounds__(256) void qkv_proj_kernel(
    const float* __restrict__ X,
    const float* __restrict__ Wq, const float* __restrict__ bq,
    const float* __restrict__ Wk, const float* __restrict__ bk,
    const float* __restrict__ Wv, const float* __restrict__ bv,
    __bf16* __restrict__ qd, __bf16* __restrict__ kd, __bf16* __restrict__ vtd)
{
  constexpr int LDA = 40, LDW = 40;   // 80B pitch: 16B-aligned, bank-spread
  __shared__ __bf16 Xl[64 * LDA];
  __shared__ __bf16 Wl[128 * LDW];

  const int tid  = threadIdx.x;
  const int m0   = blockIdx.x * 64;
  const int nblk = blockIdx.y;             // 24 blocks over 3*1024 columns
  const int p     = (nblk * 128) >> 10;    // 0=q, 1=k, 2=v
  const int n_in0 = (nblk * 128) & 1023;

  const float* W    = (p == 0) ? Wq : (p == 1) ? Wk : Wv;
  const float* bias = (p == 0) ? bq : (p == 1) ? bk : bv;

  const int wave = tid >> 5;
  const int lane = tid & 31;
  const int wm = (wave >> 2) * 32;         // 0 or 32
  const int wn = (wave & 3) * 32;          // 0..96
  const int n16 = lane & 15;
  const int mb  = (lane >> 4) * 8;

  // per-thread staging coordinates (same for load and store phases)
  const int xr_r = tid >> 3,          xr_c = (tid & 7) * 4;        // +row 0/32
  const int wr_r = tid >> 3,          wr_c = (tid & 7) * 4;        // +row 0..96

  v8f zero = {0.f,0.f,0.f,0.f,0.f,0.f,0.f,0.f};
  v8f acc[2][2];
#pragma unroll
  for (int i = 0; i < 2; ++i)
#pragma unroll
    for (int j = 0; j < 2; ++j) acc[i][j] = zero;

  v4f xr[2], wr[4];
  // prologue: load chunk 0
#pragma unroll
  for (int i = 0; i < 2; ++i)
    xr[i] = *(const v4f*)&X[(m0 + xr_r + 32 * i) * E_DIM + xr_c];
#pragma unroll
  for (int i = 0; i < 4; ++i)
    wr[i] = *(const v4f*)&W[(n_in0 + wr_r + 32 * i) * E_DIM + wr_c];

  for (int k0 = 0; k0 < E_DIM; k0 += 32) {
    __syncthreads();                       // LDS free of last chunk's readers
#pragma unroll
    for (int i = 0; i < 2; ++i)
      *(v4bf*)&Xl[(xr_r + 32 * i) * LDA + xr_c] = cvt4(xr[i]);
#pragma unroll
    for (int i = 0; i < 4; ++i)
      *(v4bf*)&Wl[(wr_r + 32 * i) * LDW + wr_c] = cvt4(wr[i]);
    __syncthreads();                       // stores visible to all waves

    if (k0 + 32 < E_DIM) {                 // issue next chunk under the WMMAs
      int k1 = k0 + 32;
#pragma unroll
      for (int i = 0; i < 2; ++i)
        xr[i] = *(const v4f*)&X[(m0 + xr_r + 32 * i) * E_DIM + k1 + xr_c];
#pragma unroll
      for (int i = 0; i < 4; ++i)
        wr[i] = *(const v4f*)&W[(n_in0 + wr_r + 32 * i) * E_DIM + k1 + wr_c];
      if (k0 + 64 < E_DIM) {               // warm L2->WGP two chunks ahead
        __builtin_prefetch(&X[(m0 + (tid & 63)) * E_DIM + k0 + 64], 0, 1);
        __builtin_prefetch(&W[(n_in0 + (tid & 127)) * E_DIM + k0 + 64], 0, 1);
      }
    }

    v16bf a0 = load_frag(Xl + (wm + 0)  * LDA, LDA);
    v16bf a1 = load_frag(Xl + (wm + 16) * LDA, LDA);
    v16bf b0 = load_frag(Wl + (wn + 0)  * LDW, LDW);
    v16bf b1 = load_frag(Wl + (wn + 16) * LDW, LDW);
    acc[0][0] = wmma_bf16(a0, b0, acc[0][0]);
    acc[0][1] = wmma_bf16(a0, b1, acc[0][1]);
    acc[1][0] = wmma_bf16(a1, b0, acc[1][0]);
    acc[1][1] = wmma_bf16(a1, b1, acc[1][1]);
  }

#pragma unroll
  for (int i = 0; i < 2; ++i) {
#pragma unroll
    for (int j = 0; j < 2; ++j) {
      int nin = n_in0 + wn + 16 * j + n16;   // column within projection
      float bval = bias[nin];
      int h = nin >> 6, d = nin & 63;
#pragma unroll
      for (int r = 0; r < 8; ++r) {
        int rg = m0 + wm + 16 * i + mb + r;  // row = t*B + b
        int t = rg >> 1, b = rg & 1;
        float val = acc[i][j][r] + bval;
        if (p == 0) {
          val *= QSCALE;
          qd[(((b * H_DIM + h) * T_DIM) + t) * DH_DIM + d] = (__bf16)val;
        } else if (p == 1) {
          kd[(((b * H_DIM + h) * T_DIM) + t) * DH_DIM + d] = (__bf16)val;
        } else {
          vtd[(((b * H_DIM + h) * DH_DIM) + d) * T_DIM + t] = (__bf16)val;
        }
      }
    }
  }
}

// ---------------------------------------------------------------------------
// Kernel 2: causal flash attention.  One block per (b, h, 64 t-rows);
// 4 waves x 16 rows.  S streamed in chunks of 32 (= WMMA K), causal-trimmed.
// K/Vt staging software-pipelined: chunk ci+1 loads overlap chunk ci's
// WMMAs + softmax.  Masks computed analytically (dense mask never read).
// ---------------------------------------------------------------------------
__global__ __launch_bounds__(128) void attn_kernel(
    const __bf16* __restrict__ qd, const __bf16* __restrict__ kd,
    const __bf16* __restrict__ vtd, const unsigned char* __restrict__ kpm,
    __bf16* __restrict__ ctx)
{
  constexpr int LQ = 72, LK = 72, LV = 40, LP = 34;
  __shared__ __bf16 QL[64 * LQ];        // Q tile  [64 t][64 d]
  __shared__ __bf16 KL[32 * LK];        // K chunk [32 s][64 d]
  __shared__ __bf16 VTL[64 * LV];       // Vt chunk[64 d][32 s]
  __shared__ __bf16 PL[4 * 16 * LP];    // per-wave P transpose staging

  const int tid  = threadIdx.x;
  const int wave = tid >> 5;
  const int lane = tid & 31;
  const int n16 = lane & 15;
  const int mb  = (lane >> 4) * 8;

  const int tile = blockIdx.x & 31;     // T/64 = 32 tiles
  const int bh   = blockIdx.x >> 5;
  const int h = bh & 15;
  const int b = bh >> 4;
  const int t0 = tile * 64;

  const __bf16* qbase = qd  + (size_t)(b * H_DIM + h) * T_DIM * DH_DIM;
  const __bf16* kbase = kd  + (size_t)(b * H_DIM + h) * T_DIM * DH_DIM;
  const __bf16* vbase = vtd + (size_t)(b * H_DIM + h) * DH_DIM * T_DIM;

#pragma unroll
  for (int i = 0; i < 4; ++i) {         // Q tile 64x64: 4x v8bf/thread
    int vi = tid + i * 128;             // 0..511
    int r = vi >> 3, c = (vi & 7) * 8;
    *(v8bf*)&QL[r * LQ + c] = *(const v8bf*)&qbase[(t0 + r) * DH_DIM + c];
  }

  v8f zero = {0.f,0.f,0.f,0.f,0.f,0.f,0.f,0.f};
  v8f O[4];
#pragma unroll
  for (int j = 0; j < 4; ++j) O[j] = zero;
  float mrow[8], lrow[8];
#pragma unroll
  for (int r = 0; r < 8; ++r) { mrow[r] = NEGF; lrow[r] = 0.f; }

  __syncthreads();
  v16bf qa0 = load_frag(QL + wave * 16 * LQ + 0,  LQ);   // d = 0..31
  v16bf qa1 = load_frag(QL + wave * 16 * LQ + 32, LQ);   // d = 32..63

  // per-thread K/Vt staging coordinates
  const int kr_r = tid >> 3, kr_c = (tid & 7) * 8;  // K rows +0/+16
  const int vr_r = tid >> 2, vr_c = (tid & 3) * 8;  // Vt rows +0/+32

  v8bf krg[2], vrg[2];
  // prologue: load chunk 0
#pragma unroll
  for (int i = 0; i < 2; ++i)
    krg[i] = *(const v8bf*)&kbase[(kr_r + 16 * i) * DH_DIM + kr_c];
#pragma unroll
  for (int i = 0; i < 2; ++i)
    vrg[i] = *(const v8bf*)&vbase[(vr_r + 32 * i) * T_DIM + vr_c];

  const int nch = tile * 2 + 2;         // causal-trimmed chunk count
  for (int ci = 0; ci < nch; ++ci) {
    int s0 = ci * 32;
    __syncthreads();                    // KL/VTL free of last chunk's readers
#pragma unroll
    for (int i = 0; i < 2; ++i)
      *(v8bf*)&KL[(kr_r + 16 * i) * LK + kr_c] = krg[i];
#pragma unroll
    for (int i = 0; i < 2; ++i)
      *(v8bf*)&VTL[(vr_r + 32 * i) * LV + vr_c] = vrg[i];
    __syncthreads();

    if (ci + 1 < nch) {                 // next chunk in flight under compute
      int s1 = s0 + 32;
#pragma unroll
      for (int i = 0; i < 2; ++i)
        krg[i] = *(const v8bf*)&kbase[(s1 + kr_r + 16 * i) * DH_DIM + kr_c];
#pragma unroll
      for (int i = 0; i < 2; ++i)
        vrg[i] = *(const v8bf*)&vbase[(vr_r + 32 * i) * T_DIM + s1 + vr_c];
      if (ci + 2 < nch) {               // prefetch two chunks ahead
        __builtin_prefetch(&kbase[(s0 + 64 + (tid & 31)) * DH_DIM], 0, 1);
        __builtin_prefetch(&vbase[(tid & 63) * T_DIM + s0 + 64], 0, 1);
      }
    }

    // scores S[16 t][32 s] = Q * K^T (Q pre-scaled), 2 N-subtiles x 2 K-steps
    v8f sc[2];
#pragma unroll
    for (int j = 0; j < 2; ++j) {
      v16bf kb0 = load_frag(KL + (16 * j) * LK + 0,  LK);
      v16bf kb1 = load_frag(KL + (16 * j) * LK + 32, LK);
      v8f a = wmma_bf16(qa0, kb0, zero);
      sc[j]  = wmma_bf16(qa1, kb1, a);
    }

    // analytic causal + key-padding mask (finite NEG, matches jnp.where)
#pragma unroll
    for (int j = 0; j < 2; ++j) {
      int s = s0 + 16 * j + n16;
      bool pad = kpm[b * T_DIM + s] != 0;
#pragma unroll
      for (int r = 0; r < 8; ++r) {
        int t = t0 + wave * 16 + mb + r;
        if (pad || s > t) sc[j][r] = NEGF;
      }
    }

    // online softmax: row stats live replicated across each 16-lane group
    float alpha[8];
#pragma unroll
    for (int r = 0; r < 8; ++r) {
      float x = fmaxf(sc[0][r], sc[1][r]);
#pragma unroll
      for (int msk = 8; msk >= 1; msk >>= 1)
        x = fmaxf(x, __shfl_xor(x, msk, 32));
      float mn = fmaxf(mrow[r], x);
      alpha[r] = __expf(mrow[r] - mn);
      mrow[r] = mn;
      float p0 = __expf(sc[0][r] - mn);
      float p1 = __expf(sc[1][r] - mn);
      sc[0][r] = p0; sc[1][r] = p1;
      float rs = p0 + p1;
#pragma unroll
      for (int msk = 8; msk >= 1; msk >>= 1)
        rs += __shfl_xor(rs, msk, 32);
      lrow[r] = lrow[r] * alpha[r] + rs;
    }
#pragma unroll
    for (int j = 0; j < 4; ++j)
#pragma unroll
      for (int r = 0; r < 8; ++r) O[j][r] *= alpha[r];

    // P: C-layout -> A-layout via per-wave LDS tile (in-order DS, no barrier)
    __bf16* pl = PL + wave * 16 * LP;
#pragma unroll
    for (int j = 0; j < 2; ++j)
#pragma unroll
      for (int r = 0; r < 8; ++r)
        pl[(mb + r) * LP + 16 * j + n16] = (__bf16)sc[j][r];
    v16bf pa = load_frag(pl, LP);

    // O[16 t][64 d] += P * V  (B operand = Vt rows [d][s], row-major)
#pragma unroll
    for (int j = 0; j < 4; ++j) {
      v16bf vb = load_frag(VTL + (16 * j) * LV, LV);
      O[j] = wmma_bf16(pa, vb, O[j]);
    }
  }

  // normalize and write ctx (bf16) in [T][B][E] row-major for final GEMM
#pragma unroll
  for (int j = 0; j < 4; ++j) {
#pragma unroll
    for (int r = 0; r < 8; ++r) {
      int t = t0 + wave * 16 + mb + r;
      int d = 16 * j + n16;
      float val = O[j][r] / lrow[r];
      ctx[(size_t)(t * B_DIM + b) * E_DIM + h * DH_DIM + d] = (__bf16)val;
    }
  }
}

// ---------------------------------------------------------------------------
// Kernel 3: output projection.  Y = ctx * Wo^T + bo, fp32 out.  Same
// software-pipelined tile loop as kernel 1.
// ---------------------------------------------------------------------------
__global__ __launch_bounds__(256) void out_proj_kernel(
    const __bf16* __restrict__ Xc,
    const float* __restrict__ Wo, const float* __restrict__ bo,
    float* __restrict__ Y)
{
  constexpr int LDA = 40, LDW = 40;
  __shared__ __bf16 Xl[64 * LDA];
  __shared__ __bf16 Wl[128 * LDW];

  const int tid = threadIdx.x;
  const int m0 = blockIdx.x * 64;
  const int n0 = blockIdx.y * 128;

  const int wave = tid >> 5;
  const int lane = tid & 31;
  const int wm = (wave >> 2) * 32;
  const int wn = (wave & 3) * 32;
  const int n16 = lane & 15;
  const int mb  = (lane >> 4) * 8;

  const int xr_r = tid >> 2, xr_c = (tid & 3) * 8;   // ctx: 1 v8bf/thread
  const int wr_r = tid >> 3, wr_c = (tid & 7) * 4;   // Wo: 4 float4/thread

  v8f zero = {0.f,0.f,0.f,0.f,0.f,0.f,0.f,0.f};
  v8f acc[2][2];
#pragma unroll
  for (int i = 0; i < 2; ++i)
#pragma unroll
    for (int j = 0; j < 2; ++j) acc[i][j] = zero;

  v8bf xr;  v4f wr[4];
  xr = *(const v8bf*)&Xc[(size_t)(m0 + xr_r) * E_DIM + xr_c];
#pragma unroll
  for (int i = 0; i < 4; ++i)
    wr[i] = *(const v4f*)&Wo[(n0 + wr_r + 32 * i) * E_DIM + wr_c];

  for (int k0 = 0; k0 < E_DIM; k0 += 32) {
    __syncthreads();
    *(v8bf*)&Xl[xr_r * LDA + xr_c] = xr;
#pragma unroll
    for (int i = 0; i < 4; ++i)
      *(v4bf*)&Wl[(wr_r + 32 * i) * LDW + wr_c] = cvt4(wr[i]);
    __syncthreads();

    if (k0 + 32 < E_DIM) {
      int k1 = k0 + 32;
      xr = *(const v8bf*)&Xc[(size_t)(m0 + xr_r) * E_DIM + k1 + xr_c];
#pragma unroll
      for (int i = 0; i < 4; ++i)
        wr[i] = *(const v4f*)&Wo[(n0 + wr_r + 32 * i) * E_DIM + k1 + wr_c];
      if (k0 + 64 < E_DIM) {
        __builtin_prefetch(&Xc[(size_t)(m0 + (tid & 63)) * E_DIM + k0 + 64], 0, 1);
        __builtin_prefetch(&Wo[(n0 + (tid & 127)) * E_DIM + k0 + 64], 0, 1);
      }
    }

    v16bf a0 = load_frag(Xl + (wm + 0)  * LDA, LDA);
    v16bf a1 = load_frag(Xl + (wm + 16) * LDA, LDA);
    v16bf b0 = load_frag(Wl + (wn + 0)  * LDW, LDW);
    v16bf b1 = load_frag(Wl + (wn + 16) * LDW, LDW);
    acc[0][0] = wmma_bf16(a0, b0, acc[0][0]);
    acc[0][1] = wmma_bf16(a0, b1, acc[0][1]);
    acc[1][0] = wmma_bf16(a1, b0, acc[1][0]);
    acc[1][1] = wmma_bf16(a1, b1, acc[1][1]);
  }

#pragma unroll
  for (int i = 0; i < 2; ++i) {
#pragma unroll
    for (int j = 0; j < 2; ++j) {
      int n = n0 + wn + 16 * j + n16;
      float bval = bo[n];
#pragma unroll
      for (int r = 0; r < 8; ++r) {
        int rg = m0 + wm + 16 * i + mb + r;
        Y[(size_t)rg * E_DIM + n] = acc[i][j][r] + bval;
      }
    }
  }
}

// ---------------------------------------------------------------------------
extern "C" void kernel_launch(void* const* d_in, const int* in_sizes, int n_in,
                              void* d_out, int out_size, void* d_ws, size_t ws_size,
                              hipStream_t stream) {
  (void)in_sizes; (void)n_in; (void)out_size; (void)ws_size;
  const float* query        = (const float*)d_in[0];
  const unsigned char* kpm  = (const unsigned char*)d_in[1];  // bool [B,T]
  // d_in[2] = dense causal attn_mask: unused (computed analytically)
  const float* Wq = (const float*)d_in[3];
  const float* bq = (const float*)d_in[4];
  const float* Wk = (const float*)d_in[5];
  const float* bk = (const float*)d_in[6];
  const float* Wv = (const float*)d_in[7];
  const float* bv = (const float*)d_in[8];
  const float* Wo = (const float*)d_in[9];
  const float* bo = (const float*)d_in[10];

  char* ws = (char*)d_ws;
  const size_t seg = (size_t)T_DIM * B_DIM * E_DIM * sizeof(__bf16); // 8 MB
  __bf16* qd  = (__bf16*)(ws + 0 * seg);   // [b][h][t][d], pre-scaled
  __bf16* kd  = (__bf16*)(ws + 1 * seg);   // [b][h][t][d]
  __bf16* vtd = (__bf16*)(ws + 2 * seg);   // [b][h][d][t]
  __bf16* ctx = (__bf16*)(ws + 3 * seg);   // [t][b][e]

  qkv_proj_kernel<<<dim3(64, 24), 256, 0, stream>>>(
      query, Wq, bq, Wk, bk, Wv, bv, qd, kd, vtd);
  attn_kernel<<<dim3(B_DIM * H_DIM * (T_DIM / 64)), 128, 0, stream>>>(
      qd, kd, vtd, kpm, ctx);
  out_proj_kernel<<<dim3(64, 8), 256, 0, stream>>>(
      ctx, Wo, bo, (float*)d_out);
}
